// SequenceTagger_75307956568175
// MI455X (gfx1250) — matrix operand order repaired and checked by reading, not compiled, not run on hardware
//
#include <hip/hip_runtime.h>
#include <hip/hip_bf16.h>
#include <math.h>

// ---------------------------------------------------------------------------
// SequenceTagger on MI455X (gfx1250, wave32).
// All GEMM-shaped work goes through v_wmma_f32_16x16x32_f16.
// Fragment layouts follow CDNA5 ISA §7.12.2:
//   A (16x32 f16): lane L -> M=L%16; VGPR v<4: K=2v,2v+1 (+8 if L>=16),
//                  v>=4: K=16+2(v-4),+1 (+8 if L>=16)
//   B (32x16 f16): lane L -> N=L%16; half i<8: K=8(L/16)+i, i>=8: K=16+8(L/16)+(i-8)
//   C/D (16x16 f32): lane L -> N=L%16; VGPR r -> M=r+8*(L/16)
// B-fragments are pre-packed (pack_weights_kernel) into per-lane layout so the
// hot loops issue only b128 vector loads; A-tiles are staged in LDS as f16 so
// A-frag register pairs are single aligned ds_load_b32s (no cvt in the loop).
// The sequential word recurrence double-buffers its pre-gate fragments into
// LDS with GLOBAL_LOAD_ASYNC_TO_LDS_B128 (ASYNCcnt), overlapping the DMA with
// the WMMA + gate math of the current step.
// ---------------------------------------------------------------------------

typedef __attribute__((ext_vector_type(16))) _Float16 v16h;
typedef __attribute__((ext_vector_type(8)))  float    v8f;

__device__ __forceinline__ v8f wmma_f16(v16h a, v16h b, v8f c) {
  // (neg_a, A, neg_b, B, c_mod, C, reuse_a, reuse_b)
  return __builtin_amdgcn_wmma_f32_16x16x32_f16(false, a, false, b, (short)0, c,
                                                false, false);
}

// ---- async global->LDS path (guarded; falls back to direct loads) ---------
#if defined(__HIP_DEVICE_COMPILE__) && \
    __has_builtin(__builtin_amdgcn_global_load_async_to_lds_b128)
#define USE_ASYNC_LDS 1
// Builtin expects pointers to 128-bit int vectors: arg0 in addrspace(1)
// (global), arg1 in addrspace(3) (LDS) -- per hipcc's own diagnostic.
typedef int async_v4i __attribute__((vector_size(16)));
typedef __attribute__((address_space(1))) async_v4i as1_v4i;
typedef __attribute__((address_space(3))) async_v4i as3_v4i;
// INST_OFFSET is added to BOTH global and LDS addresses (ISA 08 §4.4), so a
// 32B copy is two b128 ops at imm offsets 0 and 16.
__device__ __forceinline__ void async_copy_b32B(const void* g, void* l) {
  as1_v4i* gp = (as1_v4i*)(uintptr_t)g;
  as3_v4i* lp = (as3_v4i*)(unsigned)(uintptr_t)l;  // low32 = LDS address
  __builtin_amdgcn_global_load_async_to_lds_b128(gp, lp, 0, 0);
  __builtin_amdgcn_global_load_async_to_lds_b128(gp, lp, 16, 0);
}
__device__ __forceinline__ void wait_async0() {
#if __has_builtin(__builtin_amdgcn_s_wait_asynccnt)
  __builtin_amdgcn_s_wait_asynccnt(0);
#else
  asm volatile("s_wait_asynccnt 0x0" ::: "memory");
#endif
}
#endif

// ---- branch-free transcendentals (avoid libm's exec-divergent paths) ------
__device__ __forceinline__ float fast_rcp(float x) {
#if __has_builtin(__builtin_amdgcn_rcpf)
  return __builtin_amdgcn_rcpf(x);
#else
  return 1.0f / x;
#endif
}
__device__ __forceinline__ float fast_exp2(float x) {
#if __has_builtin(__builtin_amdgcn_exp2f)
  return __builtin_amdgcn_exp2f(x);
#else
  return exp2f(x);
#endif
}
__device__ __forceinline__ float fast_tanhf(float x) {
#if __has_builtin(__builtin_amdgcn_tanhf)
  return __builtin_amdgcn_tanhf(x);   // native V_TANH_F32 on gfx1250
#else
  // tanh(x) = 1 - 2/(1+exp(2x)); 2*log2(e) = 2.8853900817779268
  return 1.0f - 2.0f * fast_rcp(1.0f + fast_exp2(x * 2.8853900817779268f));
#endif
}
__device__ __forceinline__ float fast_sigmoidf(float x) {
  return 0.5f * fast_tanhf(0.5f * x) + 0.5f;
}
// Abramowitz-Stegun 7.1.26 (|err| <= 1.5e-7), branch-free.
__device__ __forceinline__ float fast_erff(float x) {
  const float ax = fabsf(x);
  const float t = fast_rcp(1.0f + 0.3275911f * ax);
  const float y =
      t * (0.254829592f +
           t * (-0.284496736f +
                t * (1.421413741f + t * (-1.453152027f + t * 1.061405429f))));
  const float e = fast_exp2(-ax * ax * 1.4426950408889634f);
  return copysignf(1.0f - y * e, x);
}

// ---- fragment helpers ------------------------------------------------------
// A-frag from f16 LDS tile (rows m=0..15, ld halfs between rows). Adjacent
// K pairs are contiguous -> each register pair is one aligned b32 LDS load.
__device__ __forceinline__ v16h load_a_frag_h(const _Float16* A, int ld) {
  const int lane = threadIdx.x & 31;
  const int m = lane & 15;
  const int kAdd = (lane >> 4) << 3;  // 0 or 8
  v16h a;
#pragma unroll
  for (int v = 0; v < 8; ++v) {
    const int k = ((v < 4) ? (v << 1) : (16 + ((v - 4) << 1))) + kAdd;
    a[2 * v] = A[m * ld + k];
    a[2 * v + 1] = A[m * ld + k + 1];
  }
  return a;
}

// Pre-packed B-fragment: one 32B per-lane vector load.
__device__ __forceinline__ v16h load_b_packed(const v16h* __restrict__ packB,
                                              int frag) {
  return packB[frag * 32 + (threadIdx.x & 31)];
}

// C-fragment broadcast of a per-column bias.
__device__ __forceinline__ v8f bias_frag(const float* __restrict__ bias,
                                         int nBase, int Nlim) {
  const int n = nBase + (threadIdx.x & 15);
  const float v = (n < Nlim) ? bias[n] : 0.0f;
  v8f c;
#pragma unroll
  for (int r = 0; r < 8; ++r) c[r] = v;
  return c;
}

// ---- B-fragment pack-kernel frag ids --------------------------------------
// gen  : frag  0..76   (j*11+kt),  B = gen_W^T      100x332 -> pad 112x352
// wpre : frag 77..140  (dir*32 + j*4 + kt), B = w_Wih^T  128x100 -> pad 128
// wrec : frag 141..156 (dir*8 + j),         B = w_Whh^T  128x32
// charC: frag 157..164 (dir*4 + j), combined [c_Wih^T ; c_Whh^T] 64x(16+16)
#define NFRAG 165

__global__ __launch_bounds__(32) void pack_weights_kernel(
    const float* __restrict__ gen_W, const float* __restrict__ w_Wih_f,
    const float* __restrict__ w_Wih_b, const float* __restrict__ w_Whh_f,
    const float* __restrict__ w_Whh_b, const float* __restrict__ c_Wih_f,
    const float* __restrict__ c_Whh_f, const float* __restrict__ c_Wih_b,
    const float* __restrict__ c_Whh_b, v16h* __restrict__ packB) {
  const int f = blockIdx.x;
  const int lane = threadIdx.x;
  const int nl = lane & 15;
  const int k0 = (lane >> 4) << 3;
  v16h b;
  if (f < 77) {
    const int j = f / 11, kt = f - j * 11;
    const int n = j * 16 + nl;
#pragma unroll
    for (int i = 0; i < 16; ++i) {
      const int k = kt * 32 + k0 + i + (i & 8);
      b[i] = (_Float16)((n < 100 && k < 332) ? gen_W[n * 332 + k] : 0.0f);
    }
  } else if (f < 141) {
    const int g = f - 77;
    const int dir = g >> 5, r = g & 31;
    const int j = r >> 2, kt = r & 3;
    const float* W = dir ? w_Wih_b : w_Wih_f;
    const int n = j * 16 + nl;
#pragma unroll
    for (int i = 0; i < 16; ++i) {
      const int k = kt * 32 + k0 + i + (i & 8);
      b[i] = (_Float16)((k < 100) ? W[n * 100 + k] : 0.0f);
    }
  } else if (f < 157) {
    const int g = f - 141;
    const int dir = g >> 3, j = g & 7;
    const float* W = dir ? w_Whh_b : w_Whh_f;
    const int n = j * 16 + nl;
#pragma unroll
    for (int i = 0; i < 16; ++i) b[i] = (_Float16)W[n * 32 + k0 + i + (i & 8)];
  } else {
    const int g = f - 157;
    const int dir = g >> 2, j = g & 3;
    const float* Wih = dir ? c_Wih_b : c_Wih_f;
    const float* Whh = dir ? c_Whh_b : c_Whh_f;
    const int n = j * 16 + nl;
#pragma unroll
    for (int i = 0; i < 16; ++i) {
      const int k = k0 + i + (i & 8);
      b[i] = (_Float16)((k < 16) ? Wih[n * 16 + k] : Whh[n * 16 + (k - 16)]);
    }
  }
  packB[f * 32 + lane] = b;
}

// ---------------------------------------------------------------------------
// Kernel 1: char embedding gather + char BiLSTM (W=16 steps, D=16, H=16).
// One wave per 16 sequences. x (K=0..15) and h (K=16..31) are FUSED into one
// A-fragment, B = [Wih^T ; Whh^T], so a full gate tile is ONE WMMA:
//   pre = x@Wih^T + h@Whh^T + bias  (4 WMMAs per step per direction).
// ---------------------------------------------------------------------------
__global__ __launch_bounds__(32) void char_bilstm_kernel(
    const int* __restrict__ char_ids, const float* __restrict__ char_table,
    const float* __restrict__ bF, const float* __restrict__ bB,
    const v16h* __restrict__ packB, float* __restrict__ char_repr) {
  __shared__ _Float16 Xh[16][16][16];  // [seq][t][d]
  __shared__ _Float16 Hh[16][16];      // h relayout buffer (C -> A layout)
  const int lane = threadIdx.x;
  const int seqBase = blockIdx.x * 16;

  for (int idx = lane; idx < 16 * 16 * 16; idx += 32) {
    const int si = idx >> 8, t = (idx >> 4) & 15, d = idx & 15;
    const int cid = char_ids[(seqBase + si) * 16 + t];
    Xh[si][t][d] = (_Float16)char_table[cid * 16 + d];
  }
  for (int idx = lane; idx < 256; idx += 32) Hh[idx >> 4][idx & 15] = (_Float16)0.0f;
  __syncthreads();

  const int nlane = lane & 15;
  const int m = lane & 15;
  const int mhi = (lane >> 4) << 3;
  const int kAdd = (lane >> 4) << 3;

  for (int dir = 0; dir < 2; ++dir) {
    const float* bb = dir ? bB : bF;
    v16h bc[4];
#pragma unroll
    for (int j = 0; j < 4; ++j) bc[j] = load_b_packed(packB, 157 + dir * 4 + j);

    v8f cst, hv;
#pragma unroll
    for (int r = 0; r < 8; ++r) { cst[r] = 0.0f; hv[r] = 0.0f; }

    for (int step = 0; step < 16; ++step) {
      const int t = dir ? (15 - step) : step;
      // fused A-frag: K 0..15 <- x[t], K 16..31 <- h
      const _Float16* Xrow = &Xh[m][t][0];
      const _Float16* Hrow = &Hh[m][0];
      v16h a;
#pragma unroll
      for (int v = 0; v < 4; ++v) {
        const int k = (v << 1) + kAdd;
        a[2 * v] = Xrow[k];
        a[2 * v + 1] = Xrow[k + 1];
      }
#pragma unroll
      for (int v = 4; v < 8; ++v) {
        const int k = ((v - 4) << 1) + kAdd;
        a[2 * v] = Hrow[k];
        a[2 * v + 1] = Hrow[k + 1];
      }
      v8f acc[4];
#pragma unroll
      for (int j = 0; j < 4; ++j) {
        acc[j] = bias_frag(bb, j * 16, 64);
        acc[j] = wmma_f16(a, bc[j], acc[j]);
      }
      // gates in torch order i, f, g, o (each exactly one 16-col N-tile)
#pragma unroll
      for (int r = 0; r < 8; ++r) {
        const float gi = fast_sigmoidf(acc[0][r]);
        const float gf = fast_sigmoidf(acc[1][r]);
        const float gg = fast_tanhf(acc[2][r]);
        const float go = fast_sigmoidf(acc[3][r]);
        const float cc = gf * cst[r] + gi * gg;
        cst[r] = cc;
        hv[r] = go * fast_tanhf(cc);
      }
      __syncthreads();
#pragma unroll
      for (int r = 0; r < 8; ++r) Hh[r + mhi][nlane] = (_Float16)hv[r];
      __syncthreads();
    }
    // final hidden state -> char_repr[:, dir*16 : dir*16+16]
#pragma unroll
    for (int r = 0; r < 8; ++r)
      char_repr[(seqBase + r + mhi) * 32 + dir * 16 + nlane] = hv[r];
    for (int idx = lane; idx < 256; idx += 32) Hh[idx >> 4][idx & 15] = (_Float16)0.0f;
    __syncthreads();
  }
}

// ---------------------------------------------------------------------------
// Kernel 2: gen layer: [8192, 332] @ gen_W^T (100x332) + b, exact-erf GELU.
// ---------------------------------------------------------------------------
__global__ __launch_bounds__(32) void gen_kernel(
    const int* __restrict__ word_ids, const float* __restrict__ word_table,
    const float* __restrict__ char_repr, const v16h* __restrict__ packB,
    const float* __restrict__ gen_b, float* __restrict__ word_vecs) {
  __shared__ _Float16 Ah[16][352];
  const int lane = threadIdx.x;
  const int mBase = blockIdx.x * 16;

  for (int idx = lane; idx < 16 * 352; idx += 32) {
    const int mm = idx / 352, k = idx - mm * 352;
    float v;
    if (k < 300)      v = word_table[word_ids[mBase + mm] * 300 + k];
    else if (k < 332) v = char_repr[(mBase + mm) * 32 + (k - 300)];
    else              v = 0.0f;
    Ah[mm][k] = (_Float16)v;
  }
  __syncthreads();

  const int nlane = lane & 15;
  const int mhi = (lane >> 4) << 3;

  for (int j = 0; j < 7; ++j) {
    v8f acc = bias_frag(gen_b, j * 16, 100);
#pragma unroll
    for (int kt = 0; kt < 11; ++kt) {
      const v16h a = load_a_frag_h(&Ah[0][kt * 32], 352);
      const v16h b = load_b_packed(packB, j * 11 + kt);
      acc = wmma_f16(a, b, acc);
    }
    const int n = j * 16 + nlane;
    if (n < 100) {
#pragma unroll
      for (int r = 0; r < 8; ++r) {
        const float x = acc[r];
        const float g = 0.5f * x * (1.0f + fast_erff(x * 0.70710678118654752f));
        word_vecs[(mBase + r + mhi) * 100 + n] = g;
      }
    }
  }
}

// ---------------------------------------------------------------------------
// Kernel 3a: word-LSTM pre-gates, both directions (parallel GEMM).
// Rows r = t*32 + batch. Output is stored in C-FRAGMENT layout so the
// sequential recurrence kernel reloads accumulators with raw b128 copies:
//   preP[(mtileGlobal*8 + j)*32 + lane] = v8f  (mtileGlobal = t*2 + mt)
// ---------------------------------------------------------------------------
__global__ __launch_bounds__(32) void word_pre_kernel(
    const float* __restrict__ word_vecs, const float* __restrict__ bF,
    const float* __restrict__ bB, const v16h* __restrict__ packB,
    float* __restrict__ preF, float* __restrict__ preB) {
  __shared__ _Float16 Ah[16][128];
  const int lane = threadIdx.x;
  const int rBase = blockIdx.x * 16;

  for (int idx = lane; idx < 16 * 128; idx += 32) {
    const int mm = idx >> 7, k = idx & 127;
    const int r = rBase + mm;
    const int t = r >> 5, bidx = r & 31;
    Ah[mm][k] = (_Float16)((k < 100) ? word_vecs[(bidx * 256 + t) * 100 + k] : 0.0f);
  }
  __syncthreads();

  for (int dir = 0; dir < 2; ++dir) {
    const float* bb = dir ? bB : bF;
    v8f* pre = (v8f*)(dir ? preB : preF);
    for (int j = 0; j < 8; ++j) {
      v8f acc = bias_frag(bb, j * 16, 128);
#pragma unroll
      for (int kt = 0; kt < 4; ++kt) {
        const v16h a = load_a_frag_h(&Ah[0][kt * 32], 128);
        const v16h b = load_b_packed(packB, 77 + dir * 32 + j * 4 + kt);
        acc = wmma_f16(a, b, acc);
      }
      pre[(blockIdx.x * 8 + j) * 32 + lane] = acc;  // frag layout
    }
  }
}

// ---------------------------------------------------------------------------
// Kernel 3b: word-LSTM recurrence (256 strictly sequential steps).
// Grid = 4 blocks: (dir, M-tile of 16 batch rows). 4 waves per block; wave w
// owns gate N-tiles {2w, 2w+1}; K = 32 = hidden -> one WMMA per N-tile.
// Pre-gate fragments for step t+1 are DMA'd into a ping-pong LDS buffer with
// GLOBAL_LOAD_ASYNC_TO_LDS_B128 while step t computes (ASYNCcnt pipelining).
// Cell state lives in registers; h bounces through a tiny f16 LDS buffer.
// ---------------------------------------------------------------------------
__global__ __launch_bounds__(128) void word_recur_kernel(
    const float* __restrict__ preF, const float* __restrict__ preB,
    const v16h* __restrict__ packB, float* __restrict__ lstm_out) {
  __shared__ _Float16 hbuf[16][32];
  __shared__ float gbuf[16][128];
#ifdef USE_ASYNC_LDS
  __shared__ __align__(32) float preLds[2][8][32][8];  // [buf][frag][lane][8]
#endif
  const int tid = threadIdx.x;
  const int wave = tid >> 5;
  const int lane = tid & 31;
  const int dir = blockIdx.x >> 1;
  const int mt = blockIdx.x & 1;
  const int bbase = mt * 16;
  const v8f* pre = (const v8f*)(dir ? preB : preF);

  for (int idx = tid; idx < 512; idx += 128) hbuf[idx >> 5][idx & 31] = (_Float16)0.0f;

  v16h bh[2];
#pragma unroll
  for (int q = 0; q < 2; ++q)
    bh[q] = load_b_packed(packB, 141 + dir * 8 + wave * 2 + q);

  float creg[4];
#pragma unroll
  for (int r = 0; r < 4; ++r) creg[r] = 0.0f;

  const int nlane = lane & 15;
  const int mhi = (lane >> 4) << 3;

#ifdef USE_ASYNC_LDS
  {  // prologue: DMA step 0 fragments into buffer 0
    const int t0 = dir ? 255 : 0;
#pragma unroll
    for (int q = 0; q < 2; ++q) {
      const int fi = wave * 2 + q;
      async_copy_b32B(&pre[((t0 * 2 + mt) * 8 + fi) * 32 + lane],
                      &preLds[0][fi][lane][0]);
    }
  }
  wait_async0();
#endif
  __syncthreads();

  for (int step = 0; step < 256; ++step) {
    const int t = dir ? (255 - step) : step;
#ifdef USE_ASYNC_LDS
    const int cur = step & 1;
    if (step + 1 < 256) {  // prefetch step+1 into the other buffer
      const int tn = dir ? (254 - step) : (step + 1);
#pragma unroll
      for (int q = 0; q < 2; ++q) {
        const int fi = wave * 2 + q;
        async_copy_b32B(&pre[((tn * 2 + mt) * 8 + fi) * 32 + lane],
                        &preLds[cur ^ 1][fi][lane][0]);
      }
    }
#endif
    const v16h a = load_a_frag_h(&hbuf[0][0], 32);
#pragma unroll
    for (int q = 0; q < 2; ++q) {
      const int n = (wave * 2 + q) * 16 + nlane;
#ifdef USE_ASYNC_LDS
      v8f acc = *(const v8f*)&preLds[cur][wave * 2 + q][lane][0];
#else
      v8f acc = pre[((t * 2 + mt) * 8 + wave * 2 + q) * 32 + lane];
#endif
      acc = wmma_f16(a, bh[q], acc);
#pragma unroll
      for (int r = 0; r < 8; ++r) gbuf[r + mhi][n] = acc[r];
    }
    __syncthreads();  // gates ready; all hbuf reads done
#pragma unroll
    for (int kk = 0; kk < 4; ++kk) {
      const int idx = tid + kk * 128;
      const int bi = idx >> 5, j = idx & 31;
      const float gi = fast_sigmoidf(gbuf[bi][j]);
      const float gf = fast_sigmoidf(gbuf[bi][32 + j]);
      const float gg = fast_tanhf(gbuf[bi][64 + j]);
      const float go = fast_sigmoidf(gbuf[bi][96 + j]);
      const float cc = gf * creg[kk] + gi * gg;
      creg[kk] = cc;
      const float hh = go * fast_tanhf(cc);
      hbuf[bi][j] = (_Float16)hh;
      lstm_out[((bbase + bi) * 256 + t) * 64 + dir * 32 + j] = hh;
    }
#ifdef USE_ASYNC_LDS
    wait_async0();  // next-step fragments landed in LDS
#endif
    __syncthreads();  // h updated (and DMA visible) before next step
  }
}

// ---------------------------------------------------------------------------
// Kernel 4: output head [8192, 64] @ out_W^T (2x64) + b, sigmoid. N=2 makes
// WMMA wasteful (14/16 dead columns); scalar dot per (row, label).
// ---------------------------------------------------------------------------
__global__ __launch_bounds__(256) void out_kernel(
    const float* __restrict__ lstm_out, const float* __restrict__ out_W,
    const float* __restrict__ out_b, float* __restrict__ out) {
  const int idx = blockIdx.x * blockDim.x + threadIdx.x;
  if (idx >= 8192 * 2) return;
  const int n = idx >> 1, l = idx & 1;
  float s = out_b[l];
  const float* row = &lstm_out[n * 64];
  const float* w = &out_W[l * 64];
#pragma unroll
  for (int k = 0; k < 64; ++k) s += row[k] * w[k];
  out[idx] = fast_sigmoidf(s);
}

// ---------------------------------------------------------------------------
extern "C" void kernel_launch(void* const* d_in, const int* in_sizes, int n_in,
                              void* d_out, int out_size, void* d_ws,
                              size_t ws_size, hipStream_t stream) {
  const int* word_ids = (const int*)d_in[0];
  const int* char_ids = (const int*)d_in[1];
  const float* word_table = (const float*)d_in[2];
  const float* char_table = (const float*)d_in[3];
  const float* c_Wih_f = (const float*)d_in[4];
  const float* c_Whh_f = (const float*)d_in[5];
  const float* c_b_f = (const float*)d_in[6];
  const float* c_Wih_b = (const float*)d_in[7];
  const float* c_Whh_b = (const float*)d_in[8];
  const float* c_b_b = (const float*)d_in[9];
  const float* gen_W = (const float*)d_in[10];
  const float* gen_b = (const float*)d_in[11];
  const float* w_Wih_f = (const float*)d_in[12];
  const float* w_Whh_f = (const float*)d_in[13];
  const float* w_b_f = (const float*)d_in[14];
  const float* w_Wih_b = (const float*)d_in[15];
  const float* w_Whh_b = (const float*)d_in[16];
  const float* w_b_b = (const float*)d_in[17];
  const float* out_W = (const float*)d_in[18];
  const float* out_b = (const float*)d_in[19];

  // workspace layout (floats), ~15 MB; packB first for 32B alignment.
  float* ws = (float*)d_ws;
  v16h* packB = (v16h*)ws;                          // NFRAG*32 v16h = 42240 f
  float* char_repr = ws + 42240;                    // 8192*32
  float* word_vecs = char_repr + 8192 * 32;         // 8192*100
  float* preF = word_vecs + 8192 * 100;             // 256*32*128 (frag layout)
  float* preB = preF + 256 * 32 * 128;              // 256*32*128 (frag layout)
  float* lstm_out = preB + 256 * 32 * 128;          // 8192*64

  pack_weights_kernel<<<NFRAG, 32, 0, stream>>>(
      gen_W, w_Wih_f, w_Wih_b, w_Whh_f, w_Whh_b, c_Wih_f, c_Whh_f, c_Wih_b,
      c_Whh_b, packB);
  char_bilstm_kernel<<<512, 32, 0, stream>>>(char_ids, char_table, c_b_f,
                                             c_b_b, packB, char_repr);
  gen_kernel<<<512, 32, 0, stream>>>(word_ids, word_table, char_repr, packB,
                                     gen_b, word_vecs);
  word_pre_kernel<<<512, 32, 0, stream>>>(word_vecs, w_b_f, w_b_b, packB,
                                          preF, preB);
  word_recur_kernel<<<4, 128, 0, stream>>>(preF, preB, packB, lstm_out);
  out_kernel<<<64, 256, 0, stream>>>(lstm_out, out_W, out_b, (float*)d_out);
}
